// Model_74474732912809
// MI455X (gfx1250) — matrix operand order repaired and checked by reading, not compiled, not run on hardware
//
#include <hip/hip_runtime.h>

#define B_   4
#define T_   3072
#define D_   1024
#define H_   16
#define HD_  64
#define TKP_ 1088   // key length 1025 padded to 17*64; pads auto-masked by 3k>q

typedef __attribute__((ext_vector_type(8)))  float  v8f;
typedef __attribute__((ext_vector_type(8)))  __bf16 v8bf;
typedef __attribute__((ext_vector_type(16))) __bf16 v16bf;
typedef int v4i_ __attribute__((__vector_size__(4 * sizeof(int))));

static __device__ __forceinline__ unsigned short f2bf(float f) {
    unsigned u = __float_as_uint(f);
    u += 0x7fffu + ((u >> 16) & 1u);           // round-to-nearest-even
    return (unsigned short)(u >> 16);
}

static __device__ __forceinline__ v16bf cat8(v8bf lo, v8bf hi) {
    return __builtin_shufflevector(lo, hi, 0,1,2,3,4,5,6,7,8,9,10,11,12,13,14,15);
}

// A-matrix 16x32 bf16 operand. Lane ln holds row M=ln; halves are the two
// 8-element runs at K = kofs + 8*lh and K = kofs + 16 + 8*lh (ISA A layout).
static __device__ __forceinline__ v16bf load_a_16x32(const __bf16* base, long rowStride,
                                                     int ln, int lh, long kofs) {
    const __bf16* p = base + (long)ln * rowStride + kofs + lh * 8;
    v8bf lo = *(const v8bf*)p;
    v8bf hi = *(const v8bf*)(p + 16);
    return cat8(lo, hi);
}

// B-matrix 32x16 bf16 operand. Lane ln holds column N=ln; halves are the two
// 8-element runs at K = kofs + 8*lh and K = kofs + 16 + 8*lh (matches A layout).
static __device__ __forceinline__ v16bf load_b_32x16(const __bf16* base, long rowStride,
                                                     int ln, int lh, long kofs) {
    const __bf16* p = base + (long)ln * rowStride + kofs + lh * 8;
    v8bf lo = *(const v8bf*)p;
    v8bf hi = *(const v8bf*)(p + 16);
    return cat8(lo, hi);
}

#define WMMA(a, b, c) \
    __builtin_amdgcn_wmma_f32_16x16x32_bf16(false, (a), false, (b), (short)0, (c), false, false)

// ---------------------------------------------------------- async LDS staging
#if defined(__gfx1250__) && __has_builtin(__builtin_amdgcn_global_load_async_to_lds_b128)
#define ASYNC_LDS 1
#endif

// Copy 16 bytes global -> LDS. Async (ASYNCcnt) when the builtin exists,
// else synchronous global_load_b128 + ds_store_b128.
static __device__ __forceinline__ void copy16_g2l(const __bf16* g, unsigned short* l) {
#ifdef ASYNC_LDS
    __builtin_amdgcn_global_load_async_to_lds_b128(
        (__attribute__((address_space(1))) v4i_*)(uintptr_t)g,
        (__attribute__((address_space(3))) v4i_*)(unsigned int)(uintptr_t)l,
        0, 0);
#else
    *(v8bf*)l = *(const v8bf*)g;
#endif
}

static __device__ __forceinline__ void async_wait0() {
#ifdef ASYNC_LDS
#if __has_builtin(__builtin_amdgcn_s_wait_asynccnt)
    __builtin_amdgcn_s_wait_asynccnt(0);
#else
    asm volatile("s_wait_asynccnt 0" ::: "memory");
#endif
#endif
}

// ---------------------------------------------------------------- pack kernels
__global__ void cvt_f32_bf16_k(const float* __restrict__ src, unsigned short* __restrict__ dst, long n) {
    long i = (long)blockIdx.x * blockDim.x + threadIdx.x;
    if (i < n) dst[i] = f2bf(src[i]);
}

// Wconv (O, I, 3) -> Wc[o][w*D + i]  (TN weight for the stride-3 conv-as-GEMM)
__global__ void repack_conv_k(const float* __restrict__ W, unsigned short* __restrict__ Wc) {
    long i = (long)blockIdx.x * blockDim.x + threadIdx.x;
    const long n = (long)D_ * D_ * 3;
    if (i >= n) return;
    int o  = (int)(i / (3 * D_));
    int r  = (int)(i % (3 * D_));
    int w  = r / D_;
    int ii = r % D_;
    Wc[i] = f2bf(W[((long)o * D_ + ii) * 3 + w]);
}

// k_tmp: row 0 = x[b,0,:], everything else zero (conv overwrites rows 1..1024)
__global__ void init_ktmp_k(const float* __restrict__ x, unsigned short* __restrict__ kt) {
    long i = (long)blockIdx.x * blockDim.x + threadIdx.x;
    const long n = (long)B_ * TKP_ * D_;
    if (i >= n) return;
    long b   = i / ((long)TKP_ * D_);
    long rem = i % ((long)TKP_ * D_);
    long row = rem / D_;
    long d   = rem % D_;
    kt[i] = (row == 0) ? f2bf(x[(b * T_) * D_ + d]) : (unsigned short)0;
}

// ------------------------------------------------- generic TN GEMM (bf16 WMMA)
// C[m,n] = sum_k A[m,k] * Bw[n,k].  Block = 4 waves; wave = 16 rows x 64 cols.
// The 64x32 B slice is shared by all 4 waves -> staged once per block into LDS
// via async global->LDS copies, double-buffered and overlapped with WMMA.
// cMode 0: bf16 row-major; 1: bf16 transposed (col*ldc+row); 2: f32 + bias.
#define BSTRIDE 40   // 32 halves + 8 pad: ds_load_b128 bank-conflict-free
__global__ __launch_bounds__(128) void gemm_tn(
    const __bf16* __restrict__ A, long aRowStride, long aBatchStride,
    const __bf16* __restrict__ Bw, int K,
    void* __restrict__ Cout, int cMode, long cBatchStride, long ldc,
    const float* __restrict__ bias)
{
    __shared__ unsigned short bstage[2][64 * BSTRIDE];
    const int tid = threadIdx.x;
    const int wv = tid >> 5, ln = tid & 15, lh = (tid >> 4) & 1;
    const int rowBase = blockIdx.x * 64 + wv * 16;
    const int colBase = blockIdx.y * 64;
    const long bb = blockIdx.z;

    const __bf16* Ab = A + bb * aBatchStride + (long)rowBase * aRowStride;
    const __bf16* Bb = Bw + (long)colBase * K;

    // this thread's two 16-byte stage segments: 64 rows x 4 segs = 256 segs
    const int s0 = tid * 2;
    const int n0 = s0 >> 2,       c0 = (s0 & 3) * 8;
    const int n1 = (s0 + 1) >> 2, c1 = ((s0 + 1) & 3) * 8;

    v8f acc[4] = {};
    // prologue: stage K-slice 0 into buffer 0
    copy16_g2l(Bb + (long)n0 * K + c0, &bstage[0][n0 * BSTRIDE + c0]);
    copy16_g2l(Bb + (long)n1 * K + c1, &bstage[0][n1 * BSTRIDE + c1]);

    int buf = 0;
    for (int kk = 0; kk < K; kk += 32) {
        async_wait0();      // this wave's copies into `buf` complete
        __syncthreads();    // all waves' copies visible
        if (kk + 32 < K) {  // kick off next K-slice into the other buffer
            copy16_g2l(Bb + (long)n0 * K + (kk + 32) + c0, &bstage[buf ^ 1][n0 * BSTRIDE + c0]);
            copy16_g2l(Bb + (long)n1 * K + (kk + 32) + c1, &bstage[buf ^ 1][n1 * BSTRIDE + c1]);
        }
        v16bf a = load_a_16x32(Ab, aRowStride, ln, lh, kk);
        const __bf16* Bl = (const __bf16*)&bstage[buf][0];
        acc[0] = WMMA(a, load_b_32x16(Bl +  0 * BSTRIDE, BSTRIDE, ln, lh, 0), acc[0]);
        acc[1] = WMMA(a, load_b_32x16(Bl + 16 * BSTRIDE, BSTRIDE, ln, lh, 0), acc[1]);
        acc[2] = WMMA(a, load_b_32x16(Bl + 32 * BSTRIDE, BSTRIDE, ln, lh, 0), acc[2]);
        acc[3] = WMMA(a, load_b_32x16(Bl + 48 * BSTRIDE, BSTRIDE, ln, lh, 0), acc[3]);
        __syncthreads();    // everyone done reading `buf` before it's reused
        buf ^= 1;
    }

#pragma unroll
    for (int j = 0; j < 4; ++j) {
#pragma unroll
        for (int r = 0; r < 8; ++r) {
            const int row = rowBase + r + 8 * lh;   // C-layout: M = r + 8*(lane>=16)
            const int col = colBase + j * 16 + ln;
            const float v = acc[j][r];
            if (cMode == 0) {
                ((unsigned short*)Cout + bb * cBatchStride)[(long)row * ldc + col] = f2bf(v);
            } else if (cMode == 1) {
                ((unsigned short*)Cout + bb * cBatchStride)[(long)col * ldc + row] = f2bf(v);
            } else {
                ((float*)Cout)[(long)row * ldc + col] = v + bias[col];
            }
        }
    }
}

// ------------------------------------------------------ flash attention (WMMA)
// grid (T/64, H, B); 4 waves/block; wave owns 16 q rows, HD=64.
__global__ __launch_bounds__(128) void attn_fa(
    const __bf16* __restrict__ Qb, const __bf16* __restrict__ Kb,
    const __bf16* __restrict__ Vt, unsigned short* __restrict__ Ob)
{
    __shared__ unsigned short ldsP[4][16][64];     // per-wave P tile (bf16)
    const int tid = threadIdx.x;
    const int wv = tid >> 5, ln = tid & 15, lh = (tid >> 4) & 1;
    const int qtile = blockIdx.x * 64 + wv * 16;
    const int h = blockIdx.y;
    const long b = blockIdx.z;

    const __bf16* Qrow = Qb + ((long)(b * T_) + qtile) * D_ + h * HD_;
    v16bf aq0 = load_a_16x32(Qrow, D_, ln, lh, 0);
    v16bf aq1 = load_a_16x32(Qrow, D_, ln, lh, 32);

    float mrun[8], lrun[8];
    v8f o[4] = {};
#pragma unroll
    for (int r = 0; r < 8; ++r) { mrun[r] = -3.0e38f; lrun[r] = 0.0f; }

    const int kallow = (qtile + 15) / 3;           // last unmasked key index
    const int ntiles = kallow / 64 + 1;            // <= 16, skips ~half the work

    for (int kt = 0; kt < ntiles; ++kt) {
        const int kb = kt * 64;
        v8f s[4];
#pragma unroll
        for (int j = 0; j < 4; ++j) {
            const __bf16* Krow = Kb + ((long)b * TKP_ + kb + j * 16) * D_ + h * HD_;
            v8f c = {};
            c = WMMA(aq0, load_b_32x16(Krow, D_, ln, lh, 0), c);
            c = WMMA(aq1, load_b_32x16(Krow, D_, ln, lh, 32), c);
            s[j] = c;
        }
        // scale (HD^-1/2 = 0.125) + mask (masked iff 3k >= q+0.001 <=> 3k > q)
#pragma unroll
        for (int j = 0; j < 4; ++j) {
            const int kg = kb + j * 16 + ln;
#pragma unroll
            for (int r = 0; r < 8; ++r) {
                const int qg = qtile + r + 8 * lh;
                float v = s[j][r] * 0.125f;
                if (3 * kg > qg) v -= 1.0e9f;
                s[j][r] = v;
            }
        }
        // online softmax: per-row (16-lane) max reduce, rescale running state
#pragma unroll
        for (int r = 0; r < 8; ++r) {
            float tm = fmaxf(fmaxf(s[0][r], s[1][r]), fmaxf(s[2][r], s[3][r]));
            tm = fmaxf(tm, __shfl_xor(tm, 1, 32));
            tm = fmaxf(tm, __shfl_xor(tm, 2, 32));
            tm = fmaxf(tm, __shfl_xor(tm, 4, 32));
            tm = fmaxf(tm, __shfl_xor(tm, 8, 32));
            const float mnew = fmaxf(mrun[r], tm);
            const float corr = __expf(mrun[r] - mnew);
            mrun[r] = mnew;
            lrun[r] *= corr;
            o[0][r] *= corr; o[1][r] *= corr; o[2][r] *= corr; o[3][r] *= corr;
        }
        // P = exp(S - m): partial row sums + C-layout -> LDS (row-major bf16)
        float ps[8];
#pragma unroll
        for (int r = 0; r < 8; ++r) ps[r] = 0.0f;
#pragma unroll
        for (int j = 0; j < 4; ++j) {
#pragma unroll
            for (int r = 0; r < 8; ++r) {
                const float p = __expf(s[j][r] - mrun[r]);
                ps[r] += p;
                ldsP[wv][r + 8 * lh][j * 16 + ln] = f2bf(p);
            }
        }
#pragma unroll
        for (int r = 0; r < 8; ++r) {
            float sm = ps[r];
            sm += __shfl_xor(sm, 1, 32);
            sm += __shfl_xor(sm, 2, 32);
            sm += __shfl_xor(sm, 4, 32);
            sm += __shfl_xor(sm, 8, 32);
            lrun[r] += sm;
        }
        // reload P in A layout (wave-private LDS; in-wave DS ordering)
        const __bf16* Pl = (const __bf16*)&ldsP[wv][0][0];
        v16bf p0 = load_a_16x32(Pl, 64, ln, lh, 0);
        v16bf p1 = load_a_16x32(Pl, 64, ln, lh, 32);
        // O += P @ V  (V pre-transposed: row = head dim, contiguous in kpos)
#pragma unroll
        for (int j = 0; j < 4; ++j) {
            const __bf16* Vrow = Vt + ((long)b * D_ + h * HD_ + j * 16) * TKP_;
            o[j] = WMMA(p0, load_b_32x16(Vrow, TKP_, ln, lh, kb), o[j]);
            o[j] = WMMA(p1, load_b_32x16(Vrow, TKP_, ln, lh, kb + 32), o[j]);
        }
    }
    // normalize and store bf16 (row-major, ready for the output projection)
#pragma unroll
    for (int r = 0; r < 8; ++r) {
        const float inv = 1.0f / lrun[r];
        const long row = (long)b * T_ + qtile + r + 8 * lh;
        unsigned short* op = Ob + row * D_ + h * HD_ + ln;
        op[0]  = f2bf(o[0][r] * inv);
        op[16] = f2bf(o[1][r] * inv);
        op[32] = f2bf(o[2][r] * inv);
        op[48] = f2bf(o[3][r] * inv);
    }
}

// ---------------------------------------------------------------------- launch
extern "C" void kernel_launch(void* const* d_in, const int* in_sizes, int n_in,
                              void* d_out, int out_size, void* d_ws, size_t ws_size,
                              hipStream_t stream)
{
    (void)in_sizes; (void)n_in; (void)out_size; (void)ws_size;
    const float* x   = (const float*)d_in[0];
    const float* Wq  = (const float*)d_in[1];
    const float* Wk  = (const float*)d_in[2];
    const float* Wv  = (const float*)d_in[3];
    const float* Wo  = (const float*)d_in[4];
    const float* bo  = (const float*)d_in[5];
    const float* Wcv = (const float*)d_in[6];
    float* out = (float*)d_out;

    char* ws = (char*)d_ws;
    size_t off = 0;
    auto alloc = [&](size_t sz) { void* p = ws + off; off += (sz + 255) & ~(size_t)255; return p; };
    const size_t SZ_XO = (size_t)B_ * T_ * D_ * 2;     // xb; reused as attention output
    const size_t SZ_KT = (size_t)B_ * TKP_ * D_ * 2;
    unsigned short* xb  = (unsigned short*)alloc(SZ_XO);
    unsigned short* qb  = (unsigned short*)alloc(SZ_XO);
    unsigned short* ktp = (unsigned short*)alloc(SZ_KT);
    unsigned short* kb  = (unsigned short*)alloc(SZ_KT);
    unsigned short* vt  = (unsigned short*)alloc(SZ_KT);
    unsigned short* wqb = (unsigned short*)alloc((size_t)D_ * D_ * 2);
    unsigned short* wkb = (unsigned short*)alloc((size_t)D_ * D_ * 2);
    unsigned short* wvb = (unsigned short*)alloc((size_t)D_ * D_ * 2);
    unsigned short* wob = (unsigned short*)alloc((size_t)D_ * D_ * 2);
    unsigned short* wcb = (unsigned short*)alloc((size_t)D_ * 3 * D_ * 2);

    const long nx = (long)B_ * T_ * D_;
    const long nw = (long)D_ * D_;
    cvt_f32_bf16_k<<<dim3((nx + 255) / 256), 256, 0, stream>>>(x,  xb,  nx);
    cvt_f32_bf16_k<<<dim3((nw + 255) / 256), 256, 0, stream>>>(Wq, wqb, nw);
    cvt_f32_bf16_k<<<dim3((nw + 255) / 256), 256, 0, stream>>>(Wk, wkb, nw);
    cvt_f32_bf16_k<<<dim3((nw + 255) / 256), 256, 0, stream>>>(Wv, wvb, nw);
    cvt_f32_bf16_k<<<dim3((nw + 255) / 256), 256, 0, stream>>>(Wo, wob, nw);
    repack_conv_k<<<dim3((int)((3 * nw + 255) / 256)), 256, 0, stream>>>(Wcv, wcb);
    const long nkt = (long)B_ * TKP_ * D_;
    init_ktmp_k<<<dim3((nkt + 255) / 256), 256, 0, stream>>>(x, ktp);

    // conv-as-GEMM: A row tkc = x[b, 3*tkc .. 3*tkc+2, :] (K=3072), out -> ktmp rows 1..1024
    gemm_tn<<<dim3(16, 16, B_), 128, 0, stream>>>(
        (const __bf16*)xb, 3 * D_, (long)T_ * D_, (const __bf16*)wcb, 3 * D_,
        (void*)(ktp + D_), 0, (long)TKP_ * D_, D_, nullptr);
    // Q = x @ Wq^T
    gemm_tn<<<dim3(192, 16, 1), 128, 0, stream>>>(
        (const __bf16*)xb, D_, 0, (const __bf16*)wqb, D_,
        (void*)qb, 0, 0, D_, nullptr);
    // K = ktmp @ Wk^T  (row-major)
    gemm_tn<<<dim3(17, 16, B_), 128, 0, stream>>>(
        (const __bf16*)ktp, D_, (long)TKP_ * D_, (const __bf16*)wkb, D_,
        (void*)kb, 0, (long)TKP_ * D_, D_, nullptr);
    // V = ktmp @ Wv^T, stored transposed (channel-major) for the P@V WMMA
    gemm_tn<<<dim3(17, 16, B_), 128, 0, stream>>>(
        (const __bf16*)ktp, D_, (long)TKP_ * D_, (const __bf16*)wvb, D_,
        (void*)vt, 1, (long)D_ * TKP_, TKP_, nullptr);
    // attention (writes into xb, which is dead after the projections)
    attn_fa<<<dim3(T_ / 64, H_, B_), 128, 0, stream>>>(
        (const __bf16*)qb, (const __bf16*)kb, (const __bf16*)vt, xb);
    // out = o @ Wo^T + bo (fp32)
    gemm_tn<<<dim3(192, 16, 1), 128, 0, stream>>>(
        (const __bf16*)xb, D_, 0, (const __bf16*)wob, D_,
        (void*)out, 2, 0, D_, bo);
}
